// AttentionTanh_13855564496966
// MI455X (gfx1250) — matrix operand order repaired and checked by reading, not compiled, not run on hardware
//
#include <hip/hip_runtime.h>
#include <stdint.h>
#include <stddef.h>

typedef __bf16 bf16_t;
typedef __attribute__((ext_vector_type(16))) __bf16 v16bf;
typedef __attribute__((ext_vector_type(8)))  __bf16 v8bf;
typedef __attribute__((ext_vector_type(8)))  float  v8f;
typedef __attribute__((ext_vector_type(4)))  float  v4f;
typedef __attribute__((ext_vector_type(4)))  uint32_t u32x4;
typedef __attribute__((ext_vector_type(8)))  int      i32x8;
typedef __attribute__((ext_vector_type(4)))  int      i32x4;

#define B_  8
#define S_  2048
#define F_  1024
#define U_  256
#define M_  (B_ * S_)   // 16384

static __device__ __forceinline__ bf16_t cvt_bf16(float f) {
  uint32_t u = __builtin_bit_cast(uint32_t, f);
  uint32_t r = u + 0x7FFFu + ((u >> 16) & 1u);   // round to nearest even
  uint16_t h = (uint16_t)(r >> 16);
  return __builtin_bit_cast(bf16_t, h);
}

// ---------------------------------------------------------------------------
// TDM: 2D tile load Global -> LDS.  Units are 8-byte elements (data_size=3).
// D# layout per CDNA5 ISA ch.8: group0 = {flags, lds_addr, global_addr, type},
// group1 = {data_size, tensor dims, tile dims, dim0 stride}.
// ---------------------------------------------------------------------------
static __device__ __forceinline__ void tdm_load_2d(uint32_t lds_off,
                                                   const void* gaddr,
                                                   uint32_t tensor_d0_units,
                                                   uint32_t tensor_d1,
                                                   uint32_t tile_d0_units,
                                                   uint32_t tile_d1,
                                                   uint32_t stride_units) {
  uint64_t ga = (uint64_t)(uintptr_t)gaddr;
  u32x4 g0;
  g0[0] = 1u;                                           // count=1 (user D#)
  g0[1] = lds_off;                                      // lds_addr
  g0[2] = (uint32_t)ga;                                 // global_addr[31:0]
  g0[3] = (uint32_t)((ga >> 32) & 0x1FFFFFFu) | (2u << 30);  // addr[56:32]|type=2
  i32x8 g1;
  g1[0] = (int)(3u << 16);                              // data_size = 8B
  g1[1] = (int)((tensor_d0_units & 0xFFFFu) << 16);     // tensor_dim0 lo
  g1[2] = (int)((tensor_d0_units >> 16) | ((tensor_d1 & 0xFFFFu) << 16));
  g1[3] = (int)((tensor_d1 >> 16) | (tile_d0_units << 16));
  g1[4] = (int)(tile_d1 & 0xFFFFu);                     // tile_dim1 (tile_dim2=0)
  g1[5] = (int)stride_units;                            // tensor_dim0_stride lo
  g1[6] = 0;
  g1[7] = 0;
  i32x4 z4 = {0, 0, 0, 0};
#if defined(__clang_major__) && (__clang_major__ >= 23)
  i32x8 z8 = {0, 0, 0, 0, 0, 0, 0, 0};
  __builtin_amdgcn_tensor_load_to_lds(g0, g1, z4, z4, z8, 0);
#else
  __builtin_amdgcn_tensor_load_to_lds(g0, g1, z4, z4, 0);
#endif
}

// ---------------------------------------------------------------------------
// f32 -> bf16 bulk conversion of x
// ---------------------------------------------------------------------------
__global__ __launch_bounds__(256)
void cvt_x_kernel(const float* __restrict__ x, bf16_t* __restrict__ xb, int n) {
  int i = (blockIdx.x * 256 + threadIdx.x) * 4;
  if (i + 3 < n) {
    v4f v = *(const v4f*)(x + i);
    xb[i + 0] = cvt_bf16(v[0]);
    xb[i + 1] = cvt_bf16(v[1]);
    xb[i + 2] = cvt_bf16(v[2]);
    xb[i + 3] = cvt_bf16(v[3]);
  }
}

// W [F=1024, U=256] f32  ->  Wt [U=256, F=1024] bf16 (transposed)
__global__ __launch_bounds__(256)
void cvt_wT_kernel(const float* __restrict__ W, bf16_t* __restrict__ Wt) {
  int idx = blockIdx.x * 256 + threadIdx.x;      // over 256*1024
  int u = idx >> 10;
  int k = idx & 1023;
  Wt[idx] = cvt_bf16(W[k * U_ + u]);
}

// ---------------------------------------------------------------------------
// Fused projection GEMM: {Q,K,V} = tanh(x @ W*) in ONE pass over x.
// Each wave: one 16-row tile of x, three 16x16 output tiles (3 accumulators).
// V written transposed [B, U, S] so attention B-fragments are contiguous.
// ---------------------------------------------------------------------------
__global__ __launch_bounds__(256)
void proj_kernel(const bf16_t* __restrict__ xb,
                 const bf16_t* __restrict__ wt,    // [3][U][F] transposed bf16
                 bf16_t* __restrict__ xq,
                 bf16_t* __restrict__ xk,
                 bf16_t* __restrict__ xvt) {
  const int lane = threadIdx.x & 31;
  const int wave = threadIdx.x >> 5;
  const int ll   = lane & 15;
  const int hi   = lane >> 4;
  const int m0 = blockIdx.x * 128 + wave * 16;
  const int n0 = blockIdx.y * 16;

  const bf16_t* arow  = xb + (size_t)(m0 + ll) * F_;
  const bf16_t* browq = wt + (size_t)(n0 + ll) * F_;
  const bf16_t* browk = browq + (size_t)(U_ * F_);
  const bf16_t* browv = browk + (size_t)(U_ * F_);

  v8f accq = {}, acck = {}, accv = {};
#pragma unroll 4
  for (int kb = 0; kb < F_; kb += 32) {
    v8bf a_lo = *(const v8bf*)(arow + kb + hi * 16);
    v8bf a_hi = *(const v8bf*)(arow + kb + hi * 16 + 8);
    v16bf A;
#pragma unroll
    for (int e = 0; e < 8; ++e) { A[e] = a_lo[e]; A[8 + e] = a_hi[e]; }
    v16bf Bq = *(const v16bf*)(browq + kb + hi * 16);
    v16bf Bk = *(const v16bf*)(browk + kb + hi * 16);
    v16bf Bv = *(const v16bf*)(browv + kb + hi * 16);
    accq = __builtin_amdgcn_wmma_f32_16x16x32_bf16(false, A, false, Bq,
                                                   (short)0, accq, false, false);
    acck = __builtin_amdgcn_wmma_f32_16x16x32_bf16(false, A, false, Bk,
                                                   (short)0, acck, false, false);
    accv = __builtin_amdgcn_wmma_f32_16x16x32_bf16(false, A, false, Bv,
                                                   (short)0, accv, false, false);
  }

  // Q, K row-major stores
  {
    bf16_t* dq = xq + (size_t)m0 * U_ + n0 + ll;
    bf16_t* dk = xk + (size_t)m0 * U_ + n0 + ll;
#pragma unroll
    for (int r = 0; r < 8; ++r) {
      dq[(size_t)(r + hi * 8) * U_] = cvt_bf16(tanhf(accq[r]));
      dk[(size_t)(r + hi * 8) * U_] = cvt_bf16(tanhf(acck[r]));
    }
  }
  // V transposed store: xvt[b][u][s]
  {
    const int b = m0 / S_;
    const int sbase = m0 % S_;
    bf16_t* dv = xvt + ((size_t)b * U_ + (n0 + ll)) * S_;
#pragma unroll
    for (int r = 0; r < 8; ++r)
      dv[sbase + r + hi * 8] = cvt_bf16(tanhf(accv[r]));
  }
}

// ---------------------------------------------------------------------------
// Fused flash attention with TDM-staged K/V tiles (double-buffered LDS).
// Block = 4 waves (64 queries); wave 0 drives the Tensor Data Mover.
// Per 32-key step: 16 QK WMMAs + 16 PV WMMAs, all operands from LDS.
// ---------------------------------------------------------------------------
struct AttnSmem {
  bf16_t K[2][32 * 256];    // key tile,  rows = key, cols = feature (16KB x2)
  bf16_t V[2][256 * 32];    // V^T tile,  rows = feature, cols = key (16KB x2)
  bf16_t P[4][16 * 32];     // per-wave P transpose staging (4KB)
};

__global__ __launch_bounds__(128)
void attn_kernel(const bf16_t* __restrict__ xq,
                 const bf16_t* __restrict__ xk,
                 const bf16_t* __restrict__ xvt,   // [B][U][S]
                 float* __restrict__ out) {
  __shared__ AttnSmem sm;
  const int lane = threadIdx.x & 31;
  const int wave = threadIdx.x >> 5;
  const int ll   = lane & 15;
  const int hi   = lane >> 4;
  const int b    = blockIdx.x >> 5;                 // 0..7
  const int q0   = (blockIdx.x & 31) * 64 + wave * 16;
  const float scale = 0.03125f;                     // 1/sqrt(1024)

  const bf16_t* Qbase = xq + ((size_t)b * S_ + q0) * U_;
  const bf16_t* Kbase = xk + (size_t)b * S_ * U_;
  const bf16_t* Vt    = xvt + (size_t)b * U_ * S_;

  const uint32_t koff0 = (uint32_t)(uintptr_t)&sm.K[0][0];
  const uint32_t koff1 = (uint32_t)(uintptr_t)&sm.K[1][0];
  const uint32_t voff0 = (uint32_t)(uintptr_t)&sm.V[0][0];
  const uint32_t voff1 = (uint32_t)(uintptr_t)&sm.V[1][0];

  // Preload the 8 Q A-fragments (U = 256 -> 8 k-steps of 32)
  v16bf Qf[8];
  {
    const bf16_t* qrow = Qbase + (size_t)ll * U_;
#pragma unroll
    for (int kk = 0; kk < 8; ++kk) {
      const int ub = kk * 32;
      v8bf lo = *(const v8bf*)(qrow + ub + hi * 16);
      v8bf hh = *(const v8bf*)(qrow + ub + hi * 16 + 8);
#pragma unroll
      for (int e = 0; e < 8; ++e) { Qf[kk][e] = lo[e]; Qf[kk][8 + e] = hh[e]; }
    }
  }

  v8f accv[16];
  v8f zero = {};
#pragma unroll
  for (int nt = 0; nt < 16; ++nt) accv[nt] = zero;
  float Mrun[8], Lrun[8];
#pragma unroll
  for (int r = 0; r < 8; ++r) { Mrun[r] = -1e30f; Lrun[r] = 0.f; }

  bf16_t* Pw = &sm.P[wave][0];

  // TDM prefetch of tile 0 (K tile: 32 rows x 512B; V^T tile: 256 rows x 64B)
  if (wave == 0) {
    tdm_load_2d(koff0, Kbase, /*t_d0*/64, /*t_d1*/S_, /*tile_d0*/64, /*tile_d1*/32, /*stride*/64);
    tdm_load_2d(voff0, Vt,    /*t_d0*/512, /*t_d1*/U_, /*tile_d0*/8, /*tile_d1*/256, /*stride*/512);
  }

  for (int t0 = 0; t0 < S_; t0 += 32) {
    const int cur = (t0 >> 5) & 1;
    // Kick the DMA for the next tile, then wait for the current one.
    if (wave == 0) {
      const int tn = (t0 + 32) & (S_ - 1);
      const uint32_t kn = (cur ? koff0 : koff1);
      const uint32_t vn = (cur ? voff0 : voff1);
      tdm_load_2d(kn, Kbase + (size_t)tn * U_, 64, S_, 64, 32, 64);
      tdm_load_2d(vn, Vt + tn,                 512, U_, 8, 256, 512);
      __builtin_amdgcn_s_wait_tensorcnt(2);   // current K+V pair landed
    }
    __syncthreads();                           // LDS tile `cur` visible to all

    const bf16_t* ldsK = &sm.K[cur][0];
    const bf16_t* ldsV = &sm.V[cur][0];

    // ---- scores for 32 keys: two 16x16 f32 fragments (operands from LDS)
    v8f s0 = zero, s1 = zero;
    const bf16_t* kr0 = ldsK + (size_t)ll * 256;
    const bf16_t* kr1 = ldsK + (size_t)(16 + ll) * 256;
#pragma unroll
    for (int kk = 0; kk < 8; ++kk) {
      const int ub = kk * 32;
      v16bf B0 = *(const v16bf*)(kr0 + ub + hi * 16);
      v16bf B1 = *(const v16bf*)(kr1 + ub + hi * 16);
      s0 = __builtin_amdgcn_wmma_f32_16x16x32_bf16(false, Qf[kk], false, B0,
                                                   (short)0, s0, false, false);
      s1 = __builtin_amdgcn_wmma_f32_16x16x32_bf16(false, Qf[kk], false, B1,
                                                   (short)0, s1, false, false);
    }

    // ---- online softmax (row = r + hi*8, cols spread over 16 lanes) ----
    float p0[8], p1[8], alpha[8];
#pragma unroll
    for (int r = 0; r < 8; ++r) {
      float a = s0[r] * scale;
      float c = s1[r] * scale;
      float tmax = fmaxf(a, c);
#pragma unroll
      for (int m = 1; m <= 8; m <<= 1)
        tmax = fmaxf(tmax, __shfl_xor(tmax, m, 32));
      float Mnew = fmaxf(Mrun[r], tmax);
      alpha[r] = __expf(Mrun[r] - Mnew);
      p0[r] = __expf(a - Mnew);
      p1[r] = __expf(c - Mnew);
      float rs = p0[r] + p1[r];
#pragma unroll
      for (int m = 1; m <= 8; m <<= 1)
        rs += __shfl_xor(rs, m, 32);
      Lrun[r] = Lrun[r] * alpha[r] + rs;
      Mrun[r] = Mnew;
    }
#pragma unroll
    for (int nt = 0; nt < 16; ++nt)
#pragma unroll
      for (int r = 0; r < 8; ++r) accv[nt][r] *= alpha[r];

    // ---- transpose P (D layout -> A layout) through per-wave LDS -------
#pragma unroll
    for (int r = 0; r < 8; ++r) {
      Pw[(r + hi * 8) * 32 + ll]      = cvt_bf16(p0[r]);
      Pw[(r + hi * 8) * 32 + 16 + ll] = cvt_bf16(p1[r]);
    }
    __syncthreads();
    v16bf Pa;
    {
      v8bf lo = *(const v8bf*)(Pw + ll * 32 + hi * 16);
      v8bf hh = *(const v8bf*)(Pw + ll * 32 + hi * 16 + 8);
#pragma unroll
      for (int e = 0; e < 8; ++e) { Pa[e] = lo[e]; Pa[8 + e] = hh[e]; }
    }

    // ---- O += P * V  (B-fragments contiguous from the V^T LDS tile) ----
#pragma unroll
    for (int nt = 0; nt < 16; ++nt) {
      v16bf Bv = *(const v16bf*)(ldsV + (size_t)(nt * 16 + ll) * 32 + hi * 16);
      accv[nt] = __builtin_amdgcn_wmma_f32_16x16x32_bf16(false, Pa, false, Bv,
                                                         (short)0, accv[nt],
                                                         false, false);
    }
    __syncthreads();   // everyone done with tile `cur` before it is re-filled
  }

  // ---- normalize and write f32 output ----------------------------------
  float* obase = out + ((size_t)b * S_ + q0) * U_;
#pragma unroll
  for (int r = 0; r < 8; ++r) {
    const float inv = 1.0f / Lrun[r];
    float* orow = obase + (size_t)(r + hi * 8) * U_;
#pragma unroll
    for (int nt = 0; nt < 16; ++nt)
      orow[nt * 16 + ll] = accv[nt][r] * inv;
  }
}

// ---------------------------------------------------------------------------
extern "C" void kernel_launch(void* const* d_in, const int* in_sizes, int n_in,
                              void* d_out, int out_size, void* d_ws, size_t ws_size,
                              hipStream_t stream) {
  (void)in_sizes; (void)n_in; (void)out_size; (void)ws_size;
  const float* x  = (const float*)d_in[0];
  const float* Wq = (const float*)d_in[1];
  const float* Wk = (const float*)d_in[2];
  const float* Wv = (const float*)d_in[3];
  float* out = (float*)d_out;

  // workspace layout (bf16): x | Wt(q,k,v) | xQ | xK | xVt   (~60.3 MB)
  bf16_t* xb  = (bf16_t*)d_ws;                  // [M, F]
  bf16_t* wt  = xb + (size_t)M_ * F_;           // [3, U, F]
  bf16_t* xq  = wt + (size_t)3 * U_ * F_;       // [M, U]
  bf16_t* xk  = xq + (size_t)M_ * U_;           // [M, U]
  bf16_t* xvt = xk + (size_t)M_ * U_;           // [B, U, S]

  cvt_x_kernel<<<(M_ * F_) / 1024, 256, 0, stream>>>(x, xb, M_ * F_);
  cvt_wT_kernel<<<(U_ * F_) / 256, 256, 0, stream>>>(Wq, wt);
  cvt_wT_kernel<<<(U_ * F_) / 256, 256, 0, stream>>>(Wk, wt + (size_t)U_ * F_);
  cvt_wT_kernel<<<(U_ * F_) / 256, 256, 0, stream>>>(Wv, wt + (size_t)2 * U_ * F_);

  dim3 pg(M_ / 128, U_ / 16);
  proj_kernel<<<pg, 256, 0, stream>>>(xb, wt, xq, xk, xvt);

  attn_kernel<<<B_ * (S_ / 64), 128, 0, stream>>>(xq, xk, xvt, out);
}
